// S4Block_22582938042899
// MI455X (gfx1250) — matrix operand order repaired and checked by reading, not compiled, not run on hardware
//
#include <hip/hip_runtime.h>
#include <math.h>

// ---------------------------------------------------------------------------
// S4 block, fused, for MI455X (gfx1250, wave32, WMMA).
// Shapes: BATCH=8, H=256, N=64 complex states, L=2048, KCH=2 (bidirectional).
// All matrix work (Toeplitz conv + 3 channel GEMMs) via v_wmma_f32_16x16x32_bf16.
// Round-2: 2 output tiles per wave sharing the B fragment -> 3 ds_load_b128
// per WMMA (was 4) in both the conv and the channel GEMMs; GEMMs moved to
// dynamic LDS (67.6 KB) to hold 64o x 64n tiles (gfx1250: 320 KB/WGP).
// ---------------------------------------------------------------------------

typedef __attribute__((ext_vector_type(16))) __bf16 v16bf;
typedef __attribute__((ext_vector_type(8)))  __bf16 v8bf;
typedef __attribute__((ext_vector_type(8)))  float  v8f;

#define HCH   256          // d_model
#define NST   64           // complex states
#define LSEQ  2048         // sequence length
#define NB    8            // batch
#define NFLAT (NB*LSEQ)    // 16384 flattened (b,l)

#define BAND_STRIDE 4104   // per-shift band row stride (elems), bank step 4
#define ZS_STRIDE   2056   // z row stride (elems), bank step 4
#define WK_STRIDE   264    // GEMM LDS row stride (elems), bank step 4

__device__ __forceinline__ float gelu_exact(float s) {
    return 0.5f * s * (1.0f + erff(s * 0.70710678118654752f));
}

__device__ __forceinline__ v16bf pack16(v8bf lo, v8bf hi) {
    v16bf a;
    #pragma unroll
    for (int e = 0; e < 8; ++e) { a[e] = lo[e]; a[e + 8] = hi[e]; }
    return a;
}

// ---------------------------------------------------------------------------
// Kernel 0: tt = t @ W_t^T + b_t  (8x256), and W_out/W1/W2 -> bf16.
// ---------------------------------------------------------------------------
__global__ void prep_kernel(const float* __restrict__ t, const float* __restrict__ W_t,
                            const float* __restrict__ b_t,
                            const float* __restrict__ W_out, const float* __restrict__ W1,
                            const float* __restrict__ W2,
                            float* __restrict__ ttws,
                            __bf16* __restrict__ Wo_bf, __bf16* __restrict__ W1_bf,
                            __bf16* __restrict__ W2_bf) {
    int i = blockIdx.x * 256 + threadIdx.x;
    Wo_bf[i] = (__bf16)W_out[i];
    W1_bf[i] = (__bf16)W1[i];
    W2_bf[i] = (__bf16)W2[i];
    if (i < NB * HCH) {                       // tt[b,d] = sum_c t[b,c] * W_t[d,c]
        int b = i >> 8, d = i & 255;
        float s = b_t[d];
        #pragma unroll 4
        for (int c = 0; c < HCH; ++c) s += t[b * HCH + c] * W_t[d * HCH + c];
        ttws[i] = s;
    }
}

// ---------------------------------------------------------------------------
// Kernel 1: LayerNorm over channels of u = x + tt, store z as bf16 (h,b,l).
// grid (8 b, 32 l-chunks of 64), block 256 = 64 l x 4 h-quarters.
// ---------------------------------------------------------------------------
__global__ void ln_kernel(const float* __restrict__ x, const float* __restrict__ ttws,
                          const float* __restrict__ ln_g, const float* __restrict__ ln_b,
                          __bf16* __restrict__ zbf) {
    __shared__ float redS[4][64], redQ[4][64], muS[64], rvS[64];
    int b = blockIdx.x, l0 = blockIdx.y * 64;
    int tid = threadIdx.x, lsub = tid & 63, hq = tid >> 6;
    int l = l0 + lsub;
    float s = 0.f, q = 0.f;
    #pragma unroll 4
    for (int hh = 0; hh < 64; ++hh) {
        int h = hq * 64 + hh;
        float u = x[(b * HCH + h) * LSEQ + l] + ttws[b * HCH + h];
        s += u; q += u * u;
    }
    redS[hq][lsub] = s; redQ[hq][lsub] = q;
    __syncthreads();
    if (hq == 0) {
        float ts = redS[0][lsub] + redS[1][lsub] + redS[2][lsub] + redS[3][lsub];
        float tq = redQ[0][lsub] + redQ[1][lsub] + redQ[2][lsub] + redQ[3][lsub];
        float mu = ts * (1.0f / HCH);
        float var = tq * (1.0f / HCH) - mu * mu;
        muS[lsub] = mu; rvS[lsub] = rsqrtf(var + 1e-5f);
    }
    __syncthreads();
    float mu = muS[lsub], rv = rvS[lsub];
    #pragma unroll 4
    for (int hh = 0; hh < 64; ++hh) {
        int h = hq * 64 + hh;
        float u = x[(b * HCH + h) * LSEQ + l] + ttws[b * HCH + h];
        float z = (u - mu) * rv * ln_g[h] + ln_b[h];
        zbf[h * NFLAT + b * LSEQ + l] = (__bf16)z;     // layout (h, b, l)
    }
}

// ---------------------------------------------------------------------------
// Kernel 2: S4D conv kernels k0/k1 (H x L, bf16) via two-level power tables:
// w^l = w^{64q} * w^r  (~6K transcendental trios per h instead of 128K).
// grid 256 (h), block 256.
// ---------------------------------------------------------------------------
__global__ void kgen_kernel(const float* __restrict__ log_dt, const float* __restrict__ log_A_real,
                            const float* __restrict__ A_imag, const float* __restrict__ B_re,
                            const float* __restrict__ B_im, const float* __restrict__ C_re,
                            const float* __restrict__ C_im,
                            __bf16* __restrict__ k0g, __bf16* __restrict__ k1g) {
    __shared__ float2 wq[NST * 32];   // w^{64q}
    __shared__ float2 wr[NST * 64];   // w^{r}
    __shared__ float2 cdb0[NST], cdb1[NST];
    __shared__ float dtar[NST], dtai[NST];
    int h = blockIdx.x, tid = threadIdx.x;

    if (tid < NST) {
        int n = tid;
        float dt  = expf(log_dt[h]);
        float Are = -expf(log_A_real[h * NST + n]);
        float Aim = A_imag[h * NST + n];
        float ar = dt * Are, ai = dt * Aim;
        dtar[n] = ar; dtai[n] = ai;
        // dB = B * (exp(dtA)-1)/A
        float er = expf(ar);
        float e1r = er * cosf(ai) - 1.0f, e1i = er * sinf(ai);
        float inv = 1.0f / (Are * Are + Aim * Aim);
        float qr = (e1r * Are + e1i * Aim) * inv;
        float qi = (e1i * Are - e1r * Aim) * inv;
        float Br = B_re[h * NST + n], Bi = B_im[h * NST + n];
        float dBr = Br * qr - Bi * qi, dBi = Br * qi + Bi * qr;
        float c0r = C_re[h * NST + n],             c0i = C_im[h * NST + n];
        float c1r = C_re[HCH * NST + h * NST + n], c1i = C_im[HCH * NST + h * NST + n];
        cdb0[n] = make_float2(c0r * dBr - c0i * dBi, c0r * dBi + c0i * dBr);
        cdb1[n] = make_float2(c1r * dBr - c1i * dBi, c1r * dBi + c1i * dBr);
    }
    __syncthreads();
    for (int idx = tid; idx < NST * 32; idx += 256) {
        int n = idx >> 5; float e = (float)((idx & 31) << 6);
        float m = expf(dtar[n] * e);
        wq[idx] = make_float2(m * cosf(dtai[n] * e), m * sinf(dtai[n] * e));
    }
    for (int idx = tid; idx < NST * 64; idx += 256) {
        int n = idx >> 6; float e = (float)(idx & 63);
        float m = expf(dtar[n] * e);
        wr[idx] = make_float2(m * cosf(dtai[n] * e), m * sinf(dtai[n] * e));
    }
    __syncthreads();
    #pragma unroll
    for (int li = 0; li < LSEQ / 256; ++li) {
        int l = li * 256 + tid;
        int qq = l >> 6, rr = l & 63;
        float a0 = 0.f, a1 = 0.f;
        #pragma unroll 8
        for (int n = 0; n < NST; ++n) {
            float2 A = wq[n * 32 + qq], B = wr[n * 64 + rr];
            float wre = A.x * B.x - A.y * B.y;
            float wim = A.x * B.y + A.y * B.x;
            a0 += cdb0[n].x * wre - cdb0[n].y * wim;
            a1 += cdb1[n].x * wre - cdb1[n].y * wim;
        }
        k0g[h * LSEQ + l] = (__bf16)(2.0f * a0);
        k1g[h * LSEQ + l] = (__bf16)(2.0f * a1);
    }
}

// ---------------------------------------------------------------------------
// Kernel 3: bidirectional Toeplitz conv via WMMA, fused +z*D and GELU.
//   y[b,l] = sum_m z[b,m] * T[l,m],  T[l,m] = band[2047 - l + m],
//   band   = reverse(k0) ++ k1.
// A-fragment base index satisfies base mod 8 == (7-row)&7  (static per lane),
// so 8 byte-shifted copies bandS[s][j] = band[j+s] give 16B-aligned b128 reads.
// D-columns 8..15 are dead (discarded in epilogue) -> no B-operand masking.
// Each wave owns TWO adjacent 16-l tiles sharing one B fragment per K-step:
// 6 ds_load_b128 per 2 WMMA. Dynamic LDS 96.3 KB (< 320 KB/WGP).
// grid (256 h, 8 l-groups), block 256 = 8 waves x 2 tiles x 16 l = 256 l.
// ---------------------------------------------------------------------------
__global__ void conv_kernel(const __bf16* __restrict__ k0g, const __bf16* __restrict__ k1g,
                            const __bf16* __restrict__ zbf, const float* __restrict__ Dv,
                            __bf16* __restrict__ ygbf) {
    extern __shared__ __bf16 smem[];
    __bf16* bandS = smem;                       // [8][BAND_STRIDE]
    __bf16* zs    = smem + 8 * BAND_STRIDE;     // [8][ZS_STRIDE]
    int h = blockIdx.x, tid = threadIdx.x;

    for (int i = tid; i < 8 * BAND_STRIDE; i += 256) {
        int s = i / BAND_STRIDE, j = i - s * BAND_STRIDE;
        int idx = j + s;
        __bf16 v = (__bf16)0.0f;
        if (idx < 2 * LSEQ)
            v = (idx < LSEQ) ? k0g[h * LSEQ + (LSEQ - 1 - idx)]
                             : k1g[h * LSEQ + (idx - LSEQ)];
        bandS[i] = v;
    }
    for (int i = tid; i < NB * LSEQ; i += 256) {
        int b = i >> 11, l = i & (LSEQ - 1);
        zs[b * ZS_STRIDE + l] = zbf[h * NFLAT + i];
    }
    __syncthreads();

    int wave = tid >> 5, lane = tid & 31;
    int l0 = (blockIdx.y * 8 + wave) * 32;     // this wave: l tiles [l0, l0+32)
    int row = lane & 15;                       // A: M index (l-tile row); B/D: N index (b)
    int half = lane >> 4;
    int klo = half * 8;                        // A elems 0..7  -> K = klo+e
    int kb  = half * 8;                        // B elems 0..7 -> K = kb+e, 8..15 -> K = kb+16+(e-8)
    int s   = (7 - row) & 7;                   // == (band base) mod 8, static
    const __bf16* bS = bandS + s * BAND_STRIDE;
    int jbase0 = (LSEQ - 1) - (l0 + row) + klo - s;  // tile0; + m0 -> mult of 8
    int jbase1 = jbase0 - 16;                        // tile1 (l0+16)
    const __bf16* zrow = zs + (row & 7) * ZS_STRIDE; // cols 8..15 alias 0..7 (dead)

    v8f acc0 = {0.f, 0.f, 0.f, 0.f, 0.f, 0.f, 0.f, 0.f};
    v8f acc1 = acc0;
    #pragma unroll 2
    for (int m0 = 0; m0 < LSEQ; m0 += 32) {
        v8bf blo = *(const v8bf*)(zrow + m0 + kb);
        v8bf bhi = *(const v8bf*)(zrow + m0 + kb + 16);
        v16bf bf = pack16(blo, bhi);                     // shared by both tiles
        v8bf a0l = *(const v8bf*)(bS + jbase0 + m0);
        v8bf a0h = *(const v8bf*)(bS + jbase0 + m0 + 16);
        acc0 = __builtin_amdgcn_wmma_f32_16x16x32_bf16(false, pack16(a0l, a0h),
                                                       false, bf, (short)0, acc0,
                                                       false, false);
        v8bf a1l = *(const v8bf*)(bS + jbase1 + m0);
        v8bf a1h = *(const v8bf*)(bS + jbase1 + m0 + 16);
        acc1 = __builtin_amdgcn_wmma_f32_16x16x32_bf16(false, pack16(a1l, a1h),
                                                       false, bf, (short)0, acc1,
                                                       false, false);
    }

    // D tiles: col N = lane&15 (=b), row M = v + 8*half (=l - tile base).
    if (row < NB) {
        float Dh = Dv[h];
        #pragma unroll
        for (int tile = 0; tile < 2; ++tile) {
            int lb = l0 + tile * 16 + half * 8;
            const v8f& acc = tile ? acc1 : acc0;
            v8bf pk;
            #pragma unroll
            for (int v = 0; v < 8; ++v) {
                float y = acc[v] + (float)zrow[lb + v] * Dh;
                pk[v] = (__bf16)gelu_exact(y);
            }
            *(v8bf*)(ygbf + h * NFLAT + row * LSEQ + lb) = pk;   // 16B aligned
        }
    }
}

// ---------------------------------------------------------------------------
// Kernels 4/5: channel GEMMs. M=o(256), N=(b,l)(16384), K=256.
// Block: 64 o x 64 n, 8 waves; each wave owns an o-pair (orow, orow+16)
// sharing one B fragment per K-step -> 6 ds_load_b128 per 2 WMMA.
// Dynamic LDS: Ws [64][264] + Bs [64][264] = 67.6 KB, conflict-free strides.
// ---------------------------------------------------------------------------
__device__ __forceinline__ void gemm_tile2(const __bf16* Ws, const __bf16* Bs,
                                           int orow, int ncol, int idx16, int half,
                                           v8f& acc0, v8f& acc1) {
    int klo = half * 8, kb = half * 8;
    const __bf16* w0 = Ws + (orow + idx16) * WK_STRIDE + klo;
    const __bf16* w1 = w0 + 16 * WK_STRIDE;
    const __bf16* br = Bs + (ncol + idx16) * WK_STRIDE + kb;
    #pragma unroll
    for (int kk = 0; kk < HCH; kk += 32) {
        v8bf blo = *(const v8bf*)(br + kk);
        v8bf bhi = *(const v8bf*)(br + kk + 16);
        v16bf bf = pack16(blo, bhi);
        v8bf a0l = *(const v8bf*)(w0 + kk);
        v8bf a0h = *(const v8bf*)(w0 + kk + 16);
        acc0 = __builtin_amdgcn_wmma_f32_16x16x32_bf16(false, pack16(a0l, a0h),
                                                       false, bf, (short)0, acc0,
                                                       false, false);
        v8bf a1l = *(const v8bf*)(w1 + kk);
        v8bf a1h = *(const v8bf*)(w1 + kk + 16);
        acc1 = __builtin_amdgcn_wmma_f32_16x16x32_bf16(false, pack16(a1l, a1h),
                                                       false, bf, (short)0, acc1,
                                                       false, false);
    }
}

__device__ __forceinline__ void stage_tiles(const __bf16* __restrict__ W_bf,
                                            const __bf16* __restrict__ src,
                                            __bf16* Ws, __bf16* Bs,
                                            int o0, int N0, int tid) {
    for (int i = tid; i < 64 * 256; i += 256)
        Ws[(i >> 8) * WK_STRIDE + (i & 255)] = W_bf[(o0 + (i >> 8)) * HCH + (i & 255)];
    for (int i = tid; i < 64 * 256; i += 256) {
        int n = i & 63, k = i >> 6;
        Bs[n * WK_STRIDE + k] = src[k * NFLAT + N0 + n];
    }
}

__global__ void gemm_gate_kernel(const __bf16* __restrict__ Wo_bf, const float* __restrict__ b_out,
                                 const float* __restrict__ x, const float* __restrict__ ttws,
                                 const __bf16* __restrict__ ygbf, __bf16* __restrict__ gbf) {
    extern __shared__ __bf16 smem[];
    __bf16* Ws = smem;                      // [64][WK_STRIDE]
    __bf16* Bs = smem + 64 * WK_STRIDE;     // [64][WK_STRIDE]
    int N0 = blockIdx.x * 64, o0 = blockIdx.y * 64, tid = threadIdx.x;
    stage_tiles(Wo_bf, ygbf, Ws, Bs, o0, N0, tid);
    __syncthreads();

    int wave = tid >> 5, lane = tid & 31;
    int orow = (wave >> 2) * 32, ncol = (wave & 3) * 16;
    int idx16 = lane & 15, half = lane >> 4;
    v8f acc0 = {0.f, 0.f, 0.f, 0.f, 0.f, 0.f, 0.f, 0.f};
    v8f acc1 = acc0;
    gemm_tile2(Ws, Bs, orow, ncol, idx16, half, acc0, acc1);

    int ng = N0 + ncol + idx16;                // global flattened (b,l)
    int bidx = ng >> 11, l = ng & (LSEQ - 1);
    #pragma unroll
    for (int p = 0; p < 2; ++p) {
        const v8f& acc = p ? acc1 : acc0;
        int obase = o0 + orow + p * 16 + half * 8;
        #pragma unroll
        for (int v = 0; v < 8; ++v) {
            int o = obase + v;
            float s = acc[v] + b_out[o] + x[(bidx * HCH + o) * LSEQ + l] + ttws[bidx * HCH + o];
            float g = tanhf(s) * (1.0f / (1.0f + expf(-s)));
            gbf[o * NFLAT + ng] = (__bf16)g;
        }
    }
}

__global__ void gemm_out_kernel(const __bf16* __restrict__ W_bf, const float* __restrict__ bias,
                                const float* __restrict__ x, const __bf16* __restrict__ gbf,
                                float* __restrict__ outp, int add_x) {
    extern __shared__ __bf16 smem[];
    __bf16* Ws = smem;
    __bf16* Bs = smem + 64 * WK_STRIDE;
    int N0 = blockIdx.x * 64, o0 = blockIdx.y * 64, tid = threadIdx.x;
    stage_tiles(W_bf, gbf, Ws, Bs, o0, N0, tid);
    __syncthreads();

    int wave = tid >> 5, lane = tid & 31;
    int orow = (wave >> 2) * 32, ncol = (wave & 3) * 16;
    int idx16 = lane & 15, half = lane >> 4;
    v8f acc0 = {0.f, 0.f, 0.f, 0.f, 0.f, 0.f, 0.f, 0.f};
    v8f acc1 = acc0;
    gemm_tile2(Ws, Bs, orow, ncol, idx16, half, acc0, acc1);

    int ng = N0 + ncol + idx16;
    int bidx = ng >> 11, l = ng & (LSEQ - 1);
    #pragma unroll
    for (int p = 0; p < 2; ++p) {
        const v8f& acc = p ? acc1 : acc0;
        int obase = o0 + orow + p * 16 + half * 8;
        #pragma unroll
        for (int v = 0; v < 8; ++v) {
            int o = obase + v;
            float s = acc[v] + bias[o];
            if (add_x) s += x[(bidx * HCH + o) * LSEQ + l];
            outp[(bidx * HCH + o) * LSEQ + l] = s;
        }
    }
}

// ---------------------------------------------------------------------------
extern "C" void kernel_launch(void* const* d_in, const int* in_sizes, int n_in,
                              void* d_out, int out_size, void* d_ws, size_t ws_size,
                              hipStream_t stream) {
    (void)in_sizes; (void)n_in; (void)out_size; (void)ws_size;
    const float* x          = (const float*)d_in[0];
    const float* t          = (const float*)d_in[1];
    const float* W_t        = (const float*)d_in[2];
    const float* b_t        = (const float*)d_in[3];
    const float* ln_g       = (const float*)d_in[4];
    const float* ln_b       = (const float*)d_in[5];
    const float* log_dt     = (const float*)d_in[6];
    const float* log_A_real = (const float*)d_in[7];
    const float* A_imag     = (const float*)d_in[8];
    const float* B_re       = (const float*)d_in[9];
    const float* B_im       = (const float*)d_in[10];
    const float* C_re       = (const float*)d_in[11];
    const float* C_im       = (const float*)d_in[12];
    const float* Dv         = (const float*)d_in[13];
    const float* W_out      = (const float*)d_in[14];
    const float* b_out      = (const float*)d_in[15];
    const float* W1         = (const float*)d_in[16];
    const float* b1         = (const float*)d_in[17];
    const float* W2         = (const float*)d_in[18];
    const float* b2         = (const float*)d_in[19];
    float* outp = (float*)d_out;

    // Workspace carving (~26.5 MB total).
    char* w = (char*)d_ws;
    size_t off = 0;
    auto alloc = [&](size_t bytes) { size_t r = off; off += (bytes + 511) & ~(size_t)511; return r; };
    float*  ttws  = (float*) (w + alloc((size_t)NB * HCH * 4));
    __bf16* Wo_bf = (__bf16*)(w + alloc((size_t)HCH * HCH * 2));
    __bf16* W1_bf = (__bf16*)(w + alloc((size_t)HCH * HCH * 2));
    __bf16* W2_bf = (__bf16*)(w + alloc((size_t)HCH * HCH * 2));
    __bf16* zbf   = (__bf16*)(w + alloc((size_t)HCH * NFLAT * 2));
    __bf16* k0bf  = (__bf16*)(w + alloc((size_t)HCH * LSEQ * 2));
    __bf16* k1bf  = (__bf16*)(w + alloc((size_t)HCH * LSEQ * 2));
    __bf16* ygbf  = (__bf16*)(w + alloc((size_t)HCH * NFLAT * 2));
    __bf16* gbf   = (__bf16*)(w + alloc((size_t)HCH * NFLAT * 2));

    const int conv_shm = (8 * BAND_STRIDE + NB * ZS_STRIDE) * (int)sizeof(__bf16); // 98,560 B
    const int gemm_shm = 128 * WK_STRIDE * (int)sizeof(__bf16);                    // 67,584 B
    (void)hipFuncSetAttribute((const void*)conv_kernel,
                              hipFuncAttributeMaxDynamicSharedMemorySize, conv_shm);
    (void)hipFuncSetAttribute((const void*)gemm_gate_kernel,
                              hipFuncAttributeMaxDynamicSharedMemorySize, gemm_shm);
    (void)hipFuncSetAttribute((const void*)gemm_out_kernel,
                              hipFuncAttributeMaxDynamicSharedMemorySize, gemm_shm);

    prep_kernel<<<dim3(256), dim3(256), 0, stream>>>(t, W_t, b_t, W_out, W1, W2,
                                                     ttws, Wo_bf, W1_bf, W2_bf);
    ln_kernel<<<dim3(8, 32), dim3(256), 0, stream>>>(x, ttws, ln_g, ln_b, zbf);
    kgen_kernel<<<dim3(256), dim3(256), 0, stream>>>(log_dt, log_A_real, A_imag,
                                                     B_re, B_im, C_re, C_im, k0bf, k1bf);
    conv_kernel<<<dim3(256, 8), dim3(256), conv_shm, stream>>>(k0bf, k1bf, zbf, Dv, ygbf);
    gemm_gate_kernel<<<dim3(256, 4), dim3(256), gemm_shm, stream>>>(Wo_bf, b_out, x, ttws,
                                                                    ygbf, gbf);
    gemm_out_kernel<<<dim3(256, 4), dim3(256), gemm_shm, stream>>>(W1_bf, b1, x, gbf, outp, 1);
    gemm_out_kernel<<<dim3(256, 4), dim3(256), gemm_shm, stream>>>(W2_bf, b2, x, gbf,
                                                                   outp + (size_t)NB * HCH * LSEQ, 0);
}
